// GaussianSplatModel_52338471469394
// MI455X (gfx1250) — matrix operand order repaired and checked by reading, not compiled, not run on hardware
//
#include <hip/hip_runtime.h>
#include <math.h>

typedef __attribute__((ext_vector_type(16))) _Float16 v16h;
typedef __attribute__((ext_vector_type(8)))  float    v8f;

#define WPB 8            // waves per block
#define TPB (WPB * 32)   // threads per block
#define TG  32           // gaussians per wave (two 16-row WMMA tiles)

__global__ __launch_bounds__(TPB)
void gs_forward_kernel(const float* __restrict__ positions,
                       const float* __restrict__ scales,
                       const float* __restrict__ rotations,
                       const float* __restrict__ opacity,
                       const float* __restrict__ features,
                       const float* __restrict__ w1, const float* __restrict__ b1,
                       const float* __restrict__ w2, const float* __restrict__ b2,
                       const float* __restrict__ w3, const float* __restrict__ b3,
                       float* __restrict__ out, int n)
{
    // per-wave LDS staging for activation layout conversion (C/D layout -> A layout)
    __shared__ _Float16 h1s[WPB][32 * 64];   // layer-1 activations, 4KB/wave
    __shared__ _Float16 h2s[WPB][32 * 32];   // layer-2 activations, 2KB/wave

    const int lane = threadIdx.x & 31;
    const int wib  = threadIdx.x >> 5;
    int base = (blockIdx.x * WPB + wib) * TG;
    // Clamp instead of early-return: overflow waves redundantly recompute tile 0
    // with identical values, keeping EXEC all-ones for WMMA and barriers safe.
    if (base >= n) base = 0;

    const int col = lane & 15;        // N (or M) index within a 16-wide tile
    const int hi  = lane >> 4;        // half-wave selector
    const int k0a = hi ? 8 : 0;       // A-matrix 16-bit layout: K chunk base
    const int k0b = hi ? 16 : 0;      // unused; B tiles are loaded with k0a (same k order as A)

    //================ element-wise part: one gaussian per lane ================
    {
        int g = base + lane;
        if (g >= n) g = n - 1;
        const float px = positions[g*3+0], py = positions[g*3+1], pz = positions[g*3+2];
        const float sx = __expf(scales[g*3+0]);
        const float sy = __expf(scales[g*3+1]);
        const float sz = __expf(scales[g*3+2]);
        float qw = rotations[g*4+0], qx = rotations[g*4+1];
        float qy = rotations[g*4+2], qz = rotations[g*4+3];
        const float inorm = rsqrtf(qw*qw + qx*qx + qy*qy + qz*qz);
        qw *= inorm; qx *= inorm; qy *= inorm; qz *= inorm;
        const float op = 1.0f / (1.0f + __expf(-opacity[g]));
        const float r00 = 1.0f - 2.0f*(qy*qy + qz*qz);
        const float r01 = 2.0f*(qx*qy - qw*qz);
        const float r02 = 2.0f*(qx*qz + qw*qy);
        const float r10 = 2.0f*(qx*qy + qw*qz);
        const float r11 = 1.0f - 2.0f*(qx*qx + qz*qz);
        const float r12 = 2.0f*(qy*qz - qw*qx);
        const float r20 = 2.0f*(qx*qz - qw*qy);
        const float r21 = 2.0f*(qy*qz + qw*qx);
        const float r22 = 1.0f - 2.0f*(qx*qx + qy*qy);
        // RS = R * diag(s);  cov = RS * RS^T (symmetric)
        const float m00=r00*sx, m01=r01*sy, m02=r02*sz;
        const float m10=r10*sx, m11=r11*sy, m12=r12*sz;
        const float m20=r20*sx, m21=r21*sy, m22=r22*sz;
        const float c00 = m00*m00 + m01*m01 + m02*m02;
        const float c01 = m00*m10 + m01*m11 + m02*m12;
        const float c02 = m00*m20 + m01*m21 + m02*m22;
        const float c11 = m10*m10 + m11*m11 + m12*m12;
        const float c12 = m10*m20 + m11*m21 + m12*m22;
        const float c22 = m20*m20 + m21*m21 + m22*m22;
        float* o = out + (size_t)g * 23;
        o[0]=px;  o[1]=py;  o[2]=pz;
        o[3]=sx;  o[4]=sy;  o[5]=sz;
        o[6]=qw;  o[7]=qx;  o[8]=qy;  o[9]=qz;
        o[10]=op;
        o[14]=c00; o[15]=c01; o[16]=c02;
        o[17]=c01; o[18]=c11; o[19]=c12;
        o[20]=c02; o[21]=c12; o[22]=c22;
    }

    //================ MLP via WMMA ================
    _Float16* __restrict__ h1 = h1s[wib];
    _Float16* __restrict__ h2 = h2s[wib];

    //---- Layer 1: features(32x32) @ w1(32x64) + b1, ReLU ----
    // A tiles (16-bit A layout: lane holds row M=col; halves = K {k0a..k0a+7} U {k0a+16..k0a+23})
    v16h A0, A1;
    {
        int r0 = base + col;          if (r0 >= n) r0 = n - 1;
        int r1 = base + 16 + col;     if (r1 >= n) r1 = n - 1;
        const float* f0 = features + (size_t)r0 * 32;
        const float* f1 = features + (size_t)r1 * 32;
        #pragma unroll
        for (int i = 0; i < 8; ++i) {
            A0[i]   = (_Float16)f0[k0a + i];
            A0[8+i] = (_Float16)f0[k0a + 16 + i];
            A1[i]   = (_Float16)f1[k0a + i];
            A1[8+i] = (_Float16)f1[k0a + 16 + i];
        }
    }
    #pragma unroll
    for (int nt = 0; nt < 4; ++nt) {
        // B tile (16-bit B layout: lane holds col N; halves = K {k0a..k0a+7} U {k0a+16..k0a+23})
        v16h B;
        #pragma unroll
        for (int i = 0; i < 16; ++i)
            B[i] = (_Float16)w1[(k0a + i + (i >= 8 ? 8 : 0)) * 64 + nt * 16 + col];
        const float bias = b1[nt * 16 + col];
        v8f acc0, acc1;
        #pragma unroll
        for (int r = 0; r < 8; ++r) { acc0[r] = bias; acc1[r] = bias; }
        acc0 = __builtin_amdgcn_wmma_f32_16x16x32_f16(false, A0, false, B, (short)0, acc0, false, false);
        acc1 = __builtin_amdgcn_wmma_f32_16x16x32_f16(false, A1, false, B, (short)0, acc1, false, false);
        // ReLU + stage to LDS (C/D layout: lane holds M = r + 8*hi, N = col)
        #pragma unroll
        for (int r = 0; r < 8; ++r) {
            const float v0 = fmaxf(acc0[r], 0.0f);
            const float v1 = fmaxf(acc1[r], 0.0f);
            h1[(r + 8*hi) * 64      + nt * 16 + col] = (_Float16)v0;
            h1[(16 + r + 8*hi) * 64 + nt * 16 + col] = (_Float16)v1;
        }
    }
    __syncthreads();

    //---- Layer 2: h1(32x64) @ w2(64x32) + b2, ReLU (K=64 -> two K=32 steps) ----
    #pragma unroll
    for (int mt = 0; mt < 2; ++mt) {
        const int arow = mt * 16 + col;
        v16h Ak0, Ak1;
        #pragma unroll
        for (int i = 0; i < 8; ++i) {
            Ak0[i]   = h1[arow * 64 + k0a + i];
            Ak0[8+i] = h1[arow * 64 + k0a + 16 + i];
            Ak1[i]   = h1[arow * 64 + 32 + k0a + i];
            Ak1[8+i] = h1[arow * 64 + 32 + k0a + 16 + i];
        }
        #pragma unroll
        for (int nt = 0; nt < 2; ++nt) {
            v16h B0, B1;
            #pragma unroll
            for (int i = 0; i < 16; ++i) {
                B0[i] = (_Float16)w2[(k0a + i + (i >= 8 ? 8 : 0)) * 32      + nt * 16 + col];
                B1[i] = (_Float16)w2[(32 + k0a + i + (i >= 8 ? 8 : 0)) * 32 + nt * 16 + col];
            }
            const float bias = b2[nt * 16 + col];
            v8f acc;
            #pragma unroll
            for (int r = 0; r < 8; ++r) acc[r] = bias;
            acc = __builtin_amdgcn_wmma_f32_16x16x32_f16(false, Ak0, false, B0, (short)0, acc, false, false);
            acc = __builtin_amdgcn_wmma_f32_16x16x32_f16(false, Ak1, false, B1, (short)0, acc, false, false);
            #pragma unroll
            for (int r = 0; r < 8; ++r) {
                const float v = fmaxf(acc[r], 0.0f);
                h2[(mt * 16 + r + 8*hi) * 32 + nt * 16 + col] = (_Float16)v;
            }
        }
    }
    __syncthreads();

    //---- Layer 3: h2(32x32) @ w3(32x3) + b3, sigmoid -> colors ----
    v16h B3;
    #pragma unroll
    for (int i = 0; i < 16; ++i)
        B3[i] = (col < 3) ? (_Float16)w3[(k0a + i + (i >= 8 ? 8 : 0)) * 3 + col] : (_Float16)0.0f;
    const float bias3 = (col < 3) ? b3[col] : 0.0f;

    #pragma unroll
    for (int mt = 0; mt < 2; ++mt) {
        const int arow = mt * 16 + col;
        v16h a;
        #pragma unroll
        for (int i = 0; i < 8; ++i) {
            a[i]   = h2[arow * 32 + k0a + i];
            a[8+i] = h2[arow * 32 + k0a + 16 + i];
        }
        v8f acc;
        #pragma unroll
        for (int r = 0; r < 8; ++r) acc[r] = bias3;
        acc = __builtin_amdgcn_wmma_f32_16x16x32_f16(false, a, false, B3, (short)0, acc, false, false);
        #pragma unroll
        for (int r = 0; r < 8; ++r) {
            const int g = base + mt * 16 + r + 8*hi;
            if (col < 3 && g < n)
                out[(size_t)g * 23 + 11 + col] = 1.0f / (1.0f + __expf(-acc[r]));
        }
    }
}

extern "C" void kernel_launch(void* const* d_in, const int* in_sizes, int n_in,
                              void* d_out, int out_size, void* d_ws, size_t ws_size,
                              hipStream_t stream) {
    (void)n_in; (void)d_ws; (void)ws_size; (void)out_size;
    const float* positions = (const float*)d_in[0];
    const float* scales    = (const float*)d_in[1];
    const float* rotations = (const float*)d_in[2];
    const float* opacity   = (const float*)d_in[3];
    const float* features  = (const float*)d_in[4];
    const float* w1        = (const float*)d_in[5];
    const float* b1        = (const float*)d_in[6];
    const float* w2        = (const float*)d_in[7];
    const float* b2        = (const float*)d_in[8];
    const float* w3        = (const float*)d_in[9];
    const float* b3        = (const float*)d_in[10];

    const int n = in_sizes[0] / 3;                 // positions is (N,3)
    const int tiles  = (n + TG - 1) / TG;          // one wave per 32 gaussians
    const int blocks = (tiles + WPB - 1) / WPB;

    gs_forward_kernel<<<blocks, TPB, 0, stream>>>(
        positions, scales, rotations, opacity, features,
        w1, b1, w2, b2, w3, b3, (float*)d_out, n);
}
